// GNNFeatureExtractor_50723563765918
// MI455X (gfx1250) — compile-verified, matrix operand order checked by reading, not run on hardware
//
#include <hip/hip_runtime.h>
#include <hip/hip_bf16.h>
#include <stdint.h>

// ---------------------------------------------------------------------------
// GCN (3-layer) inference for MI455X / gfx1250.
//
// Cost model: GEMMs ~13 GFLOP (negligible). Edge gather+scatter dominates:
// ~10 GB of traffic -> ~0.45 ms @ 23.3 TB/s HBM. Design:
//   - degree & scatter-add via native global_atomic_add_f32 (unsafeAtomicAdd)
//   - self-loop term written non-atomically as the Agg initializer
//   - layer-1/2 GEMMs on the WMMA pipe: v_wmma_f32_16x16x32_bf16,
//     bf16 activations + fp32 accumulate (A fits in 192MB L2, re-reads are cheap)
//   - BN/ReLU/denan fused into one elementwise pass that also emits the bf16
//     activation for the next GEMM; final layer fuses the mean-pool atomics.
// ---------------------------------------------------------------------------

typedef __attribute__((ext_vector_type(16))) __bf16 v16bf;
typedef __attribute__((ext_vector_type(8)))  __bf16 v8bf;
typedef __attribute__((ext_vector_type(4)))  __bf16 v4bf;
typedef __attribute__((ext_vector_type(8)))  float   v8f;

#define FDIM    256
#define NGRAPHS 64
#define BN_EPS  1e-5f

__device__ __forceinline__ float denanf(float x) { return (x != x) ? 0.0f : x; }

__device__ __forceinline__ float bn_eval(float a, float bb, float g, float be,
                                         float m, float v) {
    return (a + bb - m) * rsqrtf(v + BN_EPS) * g + be;
}

// ---------------- degree / norm -------------------------------------------
__global__ void k_init_deg(float* __restrict__ deg, int N) {
    int i = blockIdx.x * 256 + threadIdx.x;
    if (i < N) deg[i] = 1.0f;                     // self-loop contribution
}

__global__ void k_deg_edges(const long long* __restrict__ ei, int E,
                            float* __restrict__ deg) {
    int e = blockIdx.x * 256 + threadIdx.x;
    if (e < E) unsafeAtomicAdd(deg + (int)ei[(size_t)E + e], 1.0f);
}

__global__ void k_dinv(float* __restrict__ deg, int N) {
    int i = blockIdx.x * 256 + threadIdx.x;
    if (i < N) deg[i] = rsqrtf(deg[i]);           // deg >= 1 always (self loops)
}

// ---------------- weight prep: transpose + bf16 ---------------------------
// Wt[n*256+k] = bf16(W[k*256+n])  (so B fragments load row-wise like A)
__global__ void k_transpose_w(const float* __restrict__ W, __bf16* __restrict__ Wt) {
    int t = blockIdx.x * 256 + threadIdx.x;       // t = n*256 + k
    int n = t >> 8, k = t & 255;
    Wt[t] = (__bf16)W[k * FDIM + n];
}

__global__ void k_zero(float* __restrict__ p, int n) {
    int i = blockIdx.x * 256 + threadIdx.x;
    if (i < n) p[i] = 0.0f;
}

__global__ void k_pool_cnt(const long long* __restrict__ batch, int N,
                           float* __restrict__ cnt) {
    int i = blockIdx.x * 256 + threadIdx.x;
    if (i < N) unsafeAtomicAdd(cnt + (int)batch[i], 1.0f);
}

// ---------------- layer 0 dense GEMM (K=3, fp32) --------------------------
__global__ void k_gemm0(const float* __restrict__ x, const float* __restrict__ W0,
                        float* __restrict__ P, int N) {
    int t = blockIdx.x * 256 + threadIdx.x;       // t = n*64 + c/4
    int n = t >> 6, c = (t & 63) * 4;
    if (n >= N) return;
    float x0 = x[n * 3 + 0], x1 = x[n * 3 + 1], x2 = x[n * 3 + 2];
    float4 w0 = *(const float4*)(W0 + 0 * FDIM + c);
    float4 w1 = *(const float4*)(W0 + 1 * FDIM + c);
    float4 w2 = *(const float4*)(W0 + 2 * FDIM + c);
    float4 r;
    r.x = x0 * w0.x + x1 * w1.x + x2 * w2.x;
    r.y = x0 * w0.y + x1 * w1.y + x2 * w2.y;
    r.z = x0 * w0.z + x1 * w1.z + x2 * w2.z;
    r.w = x0 * w0.w + x1 * w1.w + x2 * w2.w;
    *(float4*)(P + (size_t)n * FDIM + c) = r;
}

// ---------------- WMMA GEMM: P[N,256] = A[N,256](bf16) @ Wt^T -------------
// One wave per 16x16 output tile, K swept in 8 steps of 32 (bf16).
// A-fragment per ISA layout: lanes 0-15 take K {0..7,16..23}, lanes 16-31
// take K {8..15,24..31} of row m0+(lane&15). B mirrors with N as the row of
// the pre-transposed weight matrix.
__global__ void k_wmma_gemm(const __bf16* __restrict__ A,
                            const __bf16* __restrict__ Wt,
                            float* __restrict__ P) {
    const int lane = threadIdx.x & 31;
    const int wave = threadIdx.x >> 5;
    const int tile = blockIdx.x * 8 + wave;       // mtiles*16 tiles total
    const int m0 = (tile >> 4) * 16;
    const int n0 = (tile & 15) * 16;
    const int lrow = lane & 15;
    const int lhi  = lane >> 4;                   // 0 or 1

    const __bf16* arow = A  + (size_t)(m0 + lrow) * FDIM + lhi * 8;
    const __bf16* brow = Wt + (size_t)(n0 + lrow) * FDIM + lhi * 8;

    v8f acc = {};
#pragma unroll
    for (int k0 = 0; k0 < FDIM; k0 += 32) {
        v8bf a_lo = *(const v8bf*)(arow + k0);
        v8bf a_hi = *(const v8bf*)(arow + k0 + 16);
        v8bf b_lo = *(const v8bf*)(brow + k0);
        v8bf b_hi = *(const v8bf*)(brow + k0 + 16);
        v16bf a = __builtin_shufflevector(a_lo, a_hi,
                      0,1,2,3,4,5,6,7,8,9,10,11,12,13,14,15);
        v16bf b = __builtin_shufflevector(b_lo, b_hi,
                      0,1,2,3,4,5,6,7,8,9,10,11,12,13,14,15);
        acc = __builtin_amdgcn_wmma_f32_16x16x32_bf16(
                  false, a, false, b, (short)0, acc, false, false);
    }
    // C/D layout: VGPR r -> (M=r, N=lane) for lanes 0-15, (M=r+8, N=lane-16)
    float* out = P + (size_t)(m0 + lhi * 8) * FDIM + n0 + lrow;
#pragma unroll
    for (int r = 0; r < 8; ++r) out[(size_t)r * FDIM] = acc[r];
}

// ---------------- self-loop init: Agg = P * dinv^2 ------------------------
__global__ void k_selfloop(const float* __restrict__ P, const float* __restrict__ dinv,
                           float* __restrict__ Agg, int N) {
    int t = blockIdx.x * 256 + threadIdx.x;
    int n = t >> 6, c = (t & 63) * 4;
    if (n >= N) return;
    float d = dinv[n], w = d * d;
    float4 v = *(const float4*)(P + (size_t)n * FDIM + c);
    float4 r; r.x = v.x * w; r.y = v.y * w; r.z = v.z * w; r.w = v.w * w;
    *(float4*)(Agg + (size_t)n * FDIM + c) = r;
}

// ---------------- edge scatter: one wave per edge -------------------------
__global__ void k_edge_scatter(const long long* __restrict__ ei, int E,
                               const float* __restrict__ dinv,
                               const float* __restrict__ P,
                               float* __restrict__ Agg) {
    int gid  = blockIdx.x * 256 + threadIdx.x;
    int e    = gid >> 5;
    int lane = gid & 31;
    if (e >= E) return;
    int src = (int)ei[e];
    int dst = (int)ei[(size_t)E + e];
    float w = dinv[src] * dinv[dst];
    const float* ps = P   + (size_t)src * FDIM;
    float*       pd = Agg + (size_t)dst * FDIM;
    int c = lane * 4;
    float4 v0 = *(const float4*)(ps + c);
    float4 v1 = *(const float4*)(ps + c + 128);
    unsafeAtomicAdd(pd + c + 0,   v0.x * w);
    unsafeAtomicAdd(pd + c + 1,   v0.y * w);
    unsafeAtomicAdd(pd + c + 2,   v0.z * w);
    unsafeAtomicAdd(pd + c + 3,   v0.w * w);
    unsafeAtomicAdd(pd + c + 128, v1.x * w);
    unsafeAtomicAdd(pd + c + 129, v1.y * w);
    unsafeAtomicAdd(pd + c + 130, v1.z * w);
    unsafeAtomicAdd(pd + c + 131, v1.w * w);
}

// ---------------- post: +bias, BN(eval), ReLU, denan -> bf16 --------------
__global__ void k_post_hidden(const float* __restrict__ Agg,
                              const float* __restrict__ b, const float* __restrict__ g,
                              const float* __restrict__ be, const float* __restrict__ m,
                              const float* __restrict__ v,
                              __bf16* __restrict__ Hbf, int N) {
    int t = blockIdx.x * 256 + threadIdx.x;
    int n = t >> 6, c = (t & 63) * 4;
    if (n >= N) return;
    float4 a = *(const float4*)(Agg + (size_t)n * FDIM + c);
    float av[4] = {a.x, a.y, a.z, a.w};
    v4bf h;
#pragma unroll
    for (int j = 0; j < 4; ++j) {
        float val = bn_eval(av[j], b[c + j], g[c + j], be[c + j], m[c + j], v[c + j]);
        val = fmaxf(val, 0.0f);                   // ReLU (NaN -> 0 either way)
        h[j] = (__bf16)denanf(val);
    }
    *(v4bf*)(Hbf + (size_t)n * FDIM + c) = h;
}

// ---------------- final post: BN, denan, fused mean-pool sum --------------
__global__ void k_post_final_pool(const float* __restrict__ Agg,
                                  const float* __restrict__ b, const float* __restrict__ g,
                                  const float* __restrict__ be, const float* __restrict__ m,
                                  const float* __restrict__ v,
                                  const long long* __restrict__ batch,
                                  float* __restrict__ pooled, int N) {
    int t = blockIdx.x * 256 + threadIdx.x;
    int n = t >> 6, c = (t & 63) * 4;
    if (n >= N) return;
    float4 a = *(const float4*)(Agg + (size_t)n * FDIM + c);
    float av[4] = {a.x, a.y, a.z, a.w};
    int gidx = (int)batch[n];
    float* pg = pooled + (size_t)gidx * FDIM;
#pragma unroll
    for (int j = 0; j < 4; ++j) {
        float val = bn_eval(av[j], b[c + j], g[c + j], be[c + j], m[c + j], v[c + j]);
        unsafeAtomicAdd(pg + c + j, denanf(val));
    }
}

__global__ void k_pool_final(const float* __restrict__ pooled,
                             const float* __restrict__ cnt,
                             float* __restrict__ out) {
    int t = blockIdx.x * 256 + threadIdx.x;       // 64*256 elements
    if (t >= NGRAPHS * FDIM) return;
    int g = t >> 8;
    out[t] = denanf(pooled[t] / fmaxf(cnt[g], 1.0f));
}

// ---------------------------------------------------------------------------
extern "C" void kernel_launch(void* const* d_in, const int* in_sizes, int n_in,
                              void* d_out, int out_size, void* d_ws, size_t ws_size,
                              hipStream_t stream) {
    const float*     x     = (const float*)d_in[0];
    const long long* ei    = (const long long*)d_in[1];
    const long long* batch = (const long long*)d_in[2];
    const float* W[3]  = {(const float*)d_in[3],  (const float*)d_in[9],  (const float*)d_in[15]};
    const float* b[3]  = {(const float*)d_in[4],  (const float*)d_in[10], (const float*)d_in[16]};
    const float* g[3]  = {(const float*)d_in[5],  (const float*)d_in[11], (const float*)d_in[17]};
    const float* be[3] = {(const float*)d_in[6],  (const float*)d_in[12], (const float*)d_in[18]};
    const float* m[3]  = {(const float*)d_in[7],  (const float*)d_in[13], (const float*)d_in[19]};
    const float* v[3]  = {(const float*)d_in[8],  (const float*)d_in[14], (const float*)d_in[20]};

    const int N = in_sizes[0] / 3;                // 50000 (divisible by 16)
    const int E = in_sizes[1] / 2;                // 1600000

    // workspace carve-up (~129 MB)
    char* ws = (char*)d_ws;
    float*  P    = (float*)ws;                                    // N*256 f32
    float*  Agg  = (float*)(ws + (size_t)N * FDIM * 4);           // N*256 f32
    __bf16* Hbf  = (__bf16*)(ws + (size_t)N * FDIM * 8);          // N*256 bf16
    float*  dinv = (float*)(ws + (size_t)N * FDIM * 10);          // N f32
    uintptr_t tp = ((uintptr_t)(ws + (size_t)N * FDIM * 10 + (size_t)N * 4) + 255)
                   & ~(uintptr_t)255;
    __bf16* Wt1    = (__bf16*)tp;                                 // 256*256 bf16
    __bf16* Wt2    = Wt1 + FDIM * FDIM;                           // 256*256 bf16
    float*  pooled = (float*)(Wt2 + FDIM * FDIM);                 // 64*256 f32
    float*  cnt    = pooled + NGRAPHS * FDIM;                     // 64 f32

    const int bN    = (N + 255) / 256;
    const int bE    = (E + 255) / 256;
    const int bNC4  = (N * 64 + 255) / 256;                       // N*256/4 threads
    const int bScat = (int)(((size_t)E * 32 + 255) / 256);        // 1 wave / edge
    const int bGemm = (N / 16) * 16 / 8;                          // tiles/8 (8 waves/blk)

    // --- normalization ---
    k_init_deg <<<bN, 256, 0, stream>>>(dinv, N);
    k_deg_edges<<<bE, 256, 0, stream>>>(ei, E, dinv);
    k_dinv     <<<bN, 256, 0, stream>>>(dinv, N);

    // --- weight prep + pool init ---
    k_transpose_w<<<FDIM * FDIM / 256, 256, 0, stream>>>(W[1], Wt1);
    k_transpose_w<<<FDIM * FDIM / 256, 256, 0, stream>>>(W[2], Wt2);
    k_zero<<<(NGRAPHS * FDIM + NGRAPHS + 255) / 256, 256, 0, stream>>>(
        pooled, NGRAPHS * FDIM + NGRAPHS);
    k_pool_cnt<<<bN, 256, 0, stream>>>(batch, N, cnt);

    // --- layer 0 ---
    k_gemm0       <<<bNC4, 256, 0, stream>>>(x, W[0], P, N);
    k_selfloop    <<<bNC4, 256, 0, stream>>>(P, dinv, Agg, N);
    k_edge_scatter<<<bScat, 256, 0, stream>>>(ei, E, dinv, P, Agg);
    k_post_hidden <<<bNC4, 256, 0, stream>>>(Agg, b[0], g[0], be[0], m[0], v[0], Hbf, N);

    // --- layer 1 (WMMA) ---
    k_wmma_gemm   <<<bGemm, 256, 0, stream>>>(Hbf, Wt1, P);
    k_selfloop    <<<bNC4, 256, 0, stream>>>(P, dinv, Agg, N);
    k_edge_scatter<<<bScat, 256, 0, stream>>>(ei, E, dinv, P, Agg);
    k_post_hidden <<<bNC4, 256, 0, stream>>>(Agg, b[1], g[1], be[1], m[1], v[1], Hbf, N);

    // --- layer 2 (WMMA) + fused pooling ---
    k_wmma_gemm     <<<bGemm, 256, 0, stream>>>(Hbf, Wt2, P);
    k_selfloop      <<<bNC4, 256, 0, stream>>>(P, dinv, Agg, N);
    k_edge_scatter  <<<bScat, 256, 0, stream>>>(ei, E, dinv, P, Agg);
    k_post_final_pool<<<bNC4, 256, 0, stream>>>(Agg, b[2], g[2], be[2], m[2], v[2],
                                                batch, pooled, N);
    k_pool_final<<<(NGRAPHS * FDIM + 255) / 256, 256, 0, stream>>>(pooled, cnt,
                                                                   (float*)d_out);
}
